// GraphAIRNet_86930138071447
// MI455X (gfx1250) — matrix operand order repaired and checked by reading, not compiled, not run on hardware
//
#include <hip/hip_runtime.h>

#define NNODES 50000
#define NEDGES 800000
#define FIN    128
#define HEADS0 4
#define C0CH   64
#define HIDC   256
#define OUTC   40
#define ETOT   (NEDGES + NNODES)

typedef __attribute__((ext_vector_type(16))) _Float16 v16h;
typedef __attribute__((ext_vector_type(8)))  float    v8f;
typedef __attribute__((ext_vector_type(4)))  unsigned u32x4;

// ---------------------------------------------------------------------------
// Fragment loads: WMMA 16x16x32 f16 operand layouts are K-contiguous per lane.
// A (16x32 MxK): lane -> row M=lane%16, half=lane/16.
//   u[v], v<4 : K = 8*half + 2v,+1      -> halves [8h .. 8h+7]      (16 B run)
//   u[v], v>=4: K = 16 + 8*half + 2(v-4) -> halves [16+8h .. 16+8h+7](16 B run)
// B (32x16 KxN): lane -> col N=lane%16, half=lane/16.
//   u[v]: same K mapping as A           -> halves [8h..8h+7], [16+8h..16+8h+7]
// With A stored row-major f16 [M][K] and B stored transposed f16 [N][K],
// every fragment is two global_load_b128 from consecutive addresses.
// ---------------------------------------------------------------------------
__device__ __forceinline__ v16h frag_A(const _Float16* ap /* row + 8*half */) {
  union { v16h v; u32x4 q[2]; } f;
  f.q[0] = *reinterpret_cast<const u32x4*>(ap);
  f.q[1] = *reinterpret_cast<const u32x4*>(ap + 16);
  return f.v;
}
__device__ __forceinline__ v16h frag_B(const _Float16* bp /* col*K + 8*half */) {
  union { v16h v; u32x4 q[2]; } f;
  f.q[0] = *reinterpret_cast<const u32x4*>(bp);
  f.q[1] = *reinterpret_cast<const u32x4*>(bp + 16);
  return f.v;
}

// ---------------- GEMM: C[M,NC] = Ah[M,K] @ Bt[NC,K]^T (+bias0 +bias1) ------
// Ah: f16 row-major [M][K]; Bt: f16 transposed [NC][K]. No LDS, no barriers.
// grid.x = M/16, grid.y = ceil(NC/64); block = 128 = 4 waves, wave = 16x16 tile.
__global__ __launch_bounds__(128) void gemm_wmma_pre(
    const _Float16* __restrict__ Ah, const _Float16* __restrict__ Bt,
    const float* __restrict__ bias0, const float* __restrict__ bias1,
    float* __restrict__ C, int K, int NC) {
  const int lane = threadIdx.x & 31;
  const int wv   = threadIdx.x >> 5;
  const int half = lane >> 4;
  const int m    = blockIdx.x * 16 + (lane & 15);
  const int n    = blockIdx.y * 64 + wv * 16 + (lane & 15);
  const int ncl  = (n < NC) ? n : (NC - 1);   // clamp: OOB cols computed, not stored

  const _Float16* ap = Ah + (size_t)m * K + 8 * half;
  const _Float16* bp = Bt + (size_t)ncl * K + 8 * half;

  v8f acc0 = {}, acc1 = {};
  for (int kc = 0; kc < K; kc += 64) {   // K is a multiple of 64 (128 or 256)
    v16h a0 = frag_A(ap + kc);
    v16h b0 = frag_B(bp + kc);
    acc0 = __builtin_amdgcn_wmma_f32_16x16x32_f16(false, a0, false, b0,
                                                  (short)0, acc0, false, false);
    v16h a1 = frag_A(ap + kc + 32);
    v16h b1 = frag_B(bp + kc + 32);
    acc1 = __builtin_amdgcn_wmma_f32_16x16x32_f16(false, a1, false, b1,
                                                  (short)0, acc1, false, false);
  }
  v8f acc = acc0 + acc1;

  if (n < NC) {
    float bb = (bias0 ? bias0[n] : 0.0f) + (bias1 ? bias1[n] : 0.0f);
    const int mrow = blockIdx.x * 16 + 8 * half;
#pragma unroll
    for (int r = 0; r < 8; ++r)
      C[(size_t)(mrow + r) * NC + (n - (lane & 15)) + (lane & 15)] = acc[r] + bb;
  }
}

// ---------------- f32 -> f16 conversions -----------------------------------
__global__ void convert_f16_kernel(const float* __restrict__ X,
                                   _Float16* __restrict__ Xh, int nTot) {
  int i = blockIdx.x * blockDim.x + threadIdx.x;
  if (i < nTot) Xh[i] = (_Float16)X[i];
}
// W f32 [K][NC] -> Wt f16 [NC][K] (transpose; coalesced reads)
__global__ void convert_transpose_kernel(const float* __restrict__ W,
                                         _Float16* __restrict__ Wt,
                                         int K, int NC) {
  int i = blockIdx.x * blockDim.x + threadIdx.x;
  if (i >= K * NC) return;
  int k = i / NC, c = i % NC;
  Wt[(size_t)c * K + k] = (_Float16)W[i];
}

// ---------------- attention dot products: a[n,h] = sum_c xl[n,h,c]*att[h,c] --
__global__ void att_dot_kernel(const float* __restrict__ xl,
                               const float* __restrict__ att_s,
                               const float* __restrict__ att_d,
                               float* __restrict__ a_s, float* __restrict__ a_d,
                               int nNodes, int Hh, int Ch) {
  int idx = blockIdx.x * blockDim.x + threadIdx.x;
  if (idx >= nNodes * Hh) return;
  int h = idx % Hh;
  const float* p = xl + (size_t)idx * Ch;  // (n*Hh+h)*Ch
  float ss = 0.f, sd = 0.f;
  for (int c = 0; c < Ch; ++c) {
    float v = p[c];
    ss += v * att_s[h * Ch + c];
    sd += v * att_d[h * Ch + c];
  }
  a_s[idx] = ss;
  a_d[idx] = sd;
}

// ---------------- ordered-uint encoding for float atomicMax -----------------
__device__ __forceinline__ unsigned fenc(float f) {
  int i = __float_as_int(f);
  return (i >= 0) ? ((unsigned)i | 0x80000000u) : ~(unsigned)i;
}
__device__ __forceinline__ float fdec(unsigned u) {
  int i = (u & 0x80000000u) ? (int)(u & 0x7FFFFFFFu) : ~(int)u;
  return __int_as_float(i);
}

__global__ void fill_u32_kernel(unsigned* __restrict__ p, unsigned v, int n) {
  int i = blockIdx.x * blockDim.x + threadIdx.x;
  if (i < n) p[i] = v;
}

// e = leaky_relu(a_src[src]+a_dst[dst]); segment-max into menc[dst,h]
__global__ void edge_logit_max_kernel(const float* __restrict__ a_s,
                                      const float* __restrict__ a_d,
                                      const int* __restrict__ src,
                                      const int* __restrict__ dst,
                                      int nE, int Hh, int total,
                                      float* __restrict__ ebuf,
                                      unsigned* __restrict__ menc) {
  int idx = blockIdx.x * blockDim.x + threadIdx.x;
  if (idx >= total) return;
  int e = idx / Hh, h = idx % Hh;
  int s, d;
  if (e < nE) { s = src[e]; d = dst[e]; } else { s = d = e - nE; }
  float v = a_s[s * Hh + h] + a_d[d * Hh + h];
  v = (v > 0.f) ? v : 0.2f * v;           // leaky_relu(0.2)
  ebuf[idx] = v;
  atomicMax(&menc[d * Hh + h], fenc(v));
}

// ex = exp(e - m[dst]); segment-sum into den[dst,h]
__global__ void edge_exp_sum_kernel(const int* __restrict__ src,
                                    const int* __restrict__ dst,
                                    int nE, int Hh, int total,
                                    float* __restrict__ ebuf,
                                    const unsigned* __restrict__ menc,
                                    float* __restrict__ den) {
  int idx = blockIdx.x * blockDim.x + threadIdx.x;
  if (idx >= total) return;
  int e = idx / Hh, h = idx % Hh;
  int d = (e < nE) ? dst[e] : (e - nE);
  float m = fdec(menc[d * Hh + h]);
  float ex = __expf(ebuf[idx] - m);
  ebuf[idx] = ex;
  atomicAdd(&den[d * Hh + h], ex);
}

// acc[dst, h, c] += xl[src, h, c] * (ex / (den+eps)); one thread per (edge,h,c)
__global__ void edge_scatter_kernel(const float* __restrict__ xl,
                                    const float* __restrict__ ebuf,
                                    const float* __restrict__ den,
                                    const int* __restrict__ src,
                                    const int* __restrict__ dst,
                                    int nE, int Hh, int Ch, long long total,
                                    float* __restrict__ acc) {
  long long idx = (long long)blockIdx.x * blockDim.x + threadIdx.x;
  if (idx >= total) return;
  int hc = Hh * Ch;
  int e = (int)(idx / hc);
  int j = (int)(idx % hc);
  int h = j / Ch;
  int s, d;
  if (e < nE) { s = src[e]; d = dst[e]; } else { s = d = e - nE; }
  float alpha = ebuf[e * Hh + h] / (den[d * Hh + h] + 1e-16f);
  atomicAdd(&acc[(size_t)d * hc + j], xl[(size_t)s * hc + j] * alpha);
}

__global__ void elu_kernel(float* __restrict__ x, int n) {
  int i = blockIdx.x * blockDim.x + threadIdx.x;
  if (i >= n) return;
  float v = x[i];
  x[i] = (v > 0.f) ? v : (__expf(v) - 1.0f);
}

// ---------------------------------------------------------------------------
extern "C" void kernel_launch(void* const* d_in, const int* in_sizes, int n_in,
                              void* d_out, int out_size, void* d_ws, size_t ws_size,
                              hipStream_t stream) {
  (void)in_sizes; (void)n_in; (void)out_size; (void)ws_size;
  const float* x     = (const float*)d_in[0];
  const int*   ei    = (const int*)d_in[1];   // [2,E] int32 (JAX x64 off)
  const float* W0    = (const float*)d_in[2];
  const float* b0    = (const float*)d_in[3];
  const float* atts0 = (const float*)d_in[4];
  const float* attd0 = (const float*)d_in[5];
  const float* Wr0   = (const float*)d_in[6];
  const float* br0   = (const float*)d_in[7];
  const float* W1    = (const float*)d_in[8];
  const float* b1    = (const float*)d_in[9];
  const float* atts1 = (const float*)d_in[10];
  const float* attd1 = (const float*)d_in[11];
  const float* Wr1   = (const float*)d_in[12];
  const float* br1   = (const float*)d_in[13];
  float* out = (float*)d_out;

  const int* srcv = ei;
  const int* dstv = ei + NEDGES;

  // -------- workspace layout (~145 MB) --------
  float*    xl0  = (float*)d_ws;                                // N*256 f32 (reused as xl1)
  float*    acc  = xl0 + (size_t)NNODES * HIDC;                 // N*256 f32 (accumulator / x1)
  float*    asrc = acc + (size_t)NNODES * HIDC;                 // N*4
  float*    adst = asrc + (size_t)NNODES * HEADS0;              // N*4
  float*    den  = adst + (size_t)NNODES * HEADS0;              // N*4
  unsigned* menc = (unsigned*)(den + (size_t)NNODES * HEADS0);  // N*4
  float*    ebuf = (float*)(menc + (size_t)NNODES * HEADS0);    // ETOT*4
  _Float16* xh   = (_Float16*)(ebuf + (size_t)ETOT * HEADS0);   // N*256 f16 (xh0 then x1h)
  _Float16* W0t  = xh + (size_t)NNODES * HIDC;                  // 256*128 f16
  _Float16* Wr0t = W0t + (size_t)HIDC * FIN;                    // 256*128 f16
  _Float16* W1t  = Wr0t + (size_t)HIDC * FIN;                   // 40*256 f16
  _Float16* Wr1t = W1t + (size_t)OUTC * HIDC;                   // 40*256 f16

  // -------- one-time f16 conversion / transposition --------
  convert_f16_kernel<<<(NNODES * FIN + 255) / 256, 256, 0, stream>>>(x, xh, NNODES * FIN);
  convert_transpose_kernel<<<(FIN * HIDC + 255) / 256, 256, 0, stream>>>(W0, W0t, FIN, HIDC);
  convert_transpose_kernel<<<(FIN * HIDC + 255) / 256, 256, 0, stream>>>(Wr0, Wr0t, FIN, HIDC);
  convert_transpose_kernel<<<(HIDC * OUTC + 255) / 256, 256, 0, stream>>>(W1, W1t, HIDC, OUTC);
  convert_transpose_kernel<<<(HIDC * OUTC + 255) / 256, 256, 0, stream>>>(Wr1, Wr1t, HIDC, OUTC);

  // ================= layer 0 =================
  gemm_wmma_pre<<<dim3(NNODES / 16, HIDC / 64), 128, 0, stream>>>(
      xh, W0t, nullptr, nullptr, xl0, FIN, HIDC);
  // acc = x@Wr0 + br0 + b0  (residual + both biases pre-folded)
  gemm_wmma_pre<<<dim3(NNODES / 16, HIDC / 64), 128, 0, stream>>>(
      xh, Wr0t, br0, b0, acc, FIN, HIDC);
  att_dot_kernel<<<(NNODES * HEADS0 + 255) / 256, 256, 0, stream>>>(
      xl0, atts0, attd0, asrc, adst, NNODES, HEADS0, C0CH);
  fill_u32_kernel<<<(NNODES * 8 + 255) / 256, 256, 0, stream>>>(
      (unsigned*)den, 0u, NNODES * 8);  // zero den (f32 0.0) + menc (< any fenc)
  {
    int tot = ETOT * HEADS0;
    edge_logit_max_kernel<<<(tot + 255) / 256, 256, 0, stream>>>(
        asrc, adst, srcv, dstv, NEDGES, HEADS0, tot, ebuf, menc);
    edge_exp_sum_kernel<<<(tot + 255) / 256, 256, 0, stream>>>(
        srcv, dstv, NEDGES, HEADS0, tot, ebuf, menc, den);
    long long st = (long long)ETOT * HIDC;
    edge_scatter_kernel<<<(unsigned)((st + 255) / 256), 256, 0, stream>>>(
        xl0, ebuf, den, srcv, dstv, NEDGES, HEADS0, C0CH, st, acc);
  }
  elu_kernel<<<(NNODES * HIDC + 255) / 256, 256, 0, stream>>>(acc, NNODES * HIDC);
  // acc is now x1; convert for layer-1 WMMA GEMMs
  convert_f16_kernel<<<(NNODES * HIDC + 255) / 256, 256, 0, stream>>>(acc, xh, NNODES * HIDC);

  // ================= layer 1 =================
  float* xl1 = xl0;  // N*40 fits in N*256 region
  gemm_wmma_pre<<<dim3(NNODES / 16, 1), 128, 0, stream>>>(
      xh, W1t, nullptr, nullptr, xl1, HIDC, OUTC);
  // out = x1@Wr1 + br1 + b1
  gemm_wmma_pre<<<dim3(NNODES / 16, 1), 128, 0, stream>>>(
      xh, Wr1t, br1, b1, out, HIDC, OUTC);
  att_dot_kernel<<<(NNODES + 255) / 256, 256, 0, stream>>>(
      xl1, atts1, attd1, asrc, adst, NNODES, 1, OUTC);
  fill_u32_kernel<<<(NNODES * 8 + 255) / 256, 256, 0, stream>>>(
      (unsigned*)den, 0u, NNODES * 8);
  {
    int tot = ETOT;
    edge_logit_max_kernel<<<(tot + 255) / 256, 256, 0, stream>>>(
        asrc, adst, srcv, dstv, NEDGES, 1, tot, ebuf, menc);
    edge_exp_sum_kernel<<<(tot + 255) / 256, 256, 0, stream>>>(
        srcv, dstv, NEDGES, 1, tot, ebuf, menc, den);
    long long st = (long long)ETOT * OUTC;
    edge_scatter_kernel<<<(unsigned)((st + 255) / 256), 256, 0, stream>>>(
        xl1, ebuf, den, srcv, dstv, NEDGES, 1, OUTC, st, out);
  }
  elu_kernel<<<(NNODES * OUTC + 255) / 256, 256, 0, stream>>>(out, NNODES * OUTC);
}
